// NPairLoss_72361609003243
// MI455X (gfx1250) — matrix operand order repaired and checked by reading, not compiled, not run on hardware
//
#include <hip/hip_runtime.h>
#include <hip/hip_bf16.h>
#include <math.h>

// ---------------------------------------------------------------------------
// NPair loss, MI455X (gfx1250). S = A P^T via bf16x3-split WMMA (fp32-class
// accuracy at bf16 matrix-core rate), fused exp/row-sum epilogue.
// Round 2: 64x64 wave tile (MT=4 x NT=4), B fragments register-resident,
// loop-carried pointers so all fragment loads use constant instruction
// offsets (no per-fragment 64-bit VALU address math in the hot loop).
// ---------------------------------------------------------------------------

typedef __attribute__((ext_vector_type(16))) __bf16        v16bf;
typedef __attribute__((ext_vector_type(8)))  float         v8f;
typedef __attribute__((ext_vector_type(4)))  unsigned int  v4u;

#define NROWS   4096
#define DDIM    1024
#define L2REG   0.02f

// gemm tiling: 8 waves = 2 (M) x 4 (N); wave tile 64x64 -> block 128x256
#define MT 4
#define NT 4
#define KSTEP 32
#define BLK_M 128
#define BLK_N 256

union Frag { v16bf h; v4u u[2]; };

static __device__ __forceinline__ unsigned short bf16_rne(float x) {
    unsigned int u = __float_as_uint(x);
    unsigned int r = u + 0x7FFFu + ((u >> 16) & 1u);
    return (unsigned short)(r >> 16);
}

// ---------------- kernel 0: zero accumulators ------------------------------
__global__ void npair_init_kernel(float* __restrict__ rowsum, float* __restrict__ l2acc) {
    int t = blockIdx.x * blockDim.x + threadIdx.x;
    if (t < NROWS) rowsum[t] = 0.0f;
    if (t == 0)    *l2acc = 0.0f;
}

// ---------------- kernel 1: fp32 -> (hi,lo) bf16 split ---------------------
__global__ void npair_split_kernel(const float* __restrict__ A, const float* __restrict__ P,
                                   unsigned short* __restrict__ Ahi, unsigned short* __restrict__ Alo,
                                   unsigned short* __restrict__ Phi, unsigned short* __restrict__ Plo) {
    size_t t = (size_t)blockIdx.x * blockDim.x + threadIdx.x;   // over NROWS*DDIM
    float a = A[t];
    unsigned short ah = bf16_rne(a);
    float ahf = __uint_as_float((unsigned int)ah << 16);
    Ahi[t] = ah;
    Alo[t] = bf16_rne(a - ahf);
    float p = P[t];
    unsigned short ph = bf16_rne(p);
    float phf = __uint_as_float((unsigned int)ph << 16);
    Phi[t] = ph;
    Plo[t] = bf16_rne(p - phf);
}

// ---------------- kernel 2: exact fp32 diagonal + L2 -----------------------
__global__ __launch_bounds__(256)
void npair_diag_kernel(const float* __restrict__ A, const float* __restrict__ P,
                       float* __restrict__ diag, float* __restrict__ l2acc) {
    const int i = blockIdx.x;
    const float* a = A + (size_t)i * DDIM;
    const float* p = P + (size_t)i * DDIM;
    float dot = 0.0f, sq = 0.0f;
    for (int k = threadIdx.x; k < DDIM; k += 256) {
        float av = a[k], pv = p[k];
        dot += av * pv;
        sq  += av * av + pv * pv;
    }
    __shared__ float sdot[256];
    __shared__ float ssq[256];
    sdot[threadIdx.x] = dot; ssq[threadIdx.x] = sq;
    __syncthreads();
    for (int s = 128; s > 0; s >>= 1) {
        if (threadIdx.x < s) { sdot[threadIdx.x] += sdot[threadIdx.x + s];
                               ssq[threadIdx.x]  += ssq[threadIdx.x + s]; }
        __syncthreads();
    }
    if (threadIdx.x == 0) { diag[i] = sdot[0]; atomicAdd(l2acc, ssq[0]); }
}

// ---------------- kernel 3: WMMA GEMM + fused exp/row-sum ------------------
// Fragment layouts used here (wave32; contiguous k order, same in A and B, not the ISA table order):
//  A (16x32 bf16): lane L<16 holds M=L, K=k0..k0+15;
//                  lane L>=16 holds M=L-16, K=k0+16..k0+31.  -> two 16B loads.
//  B (32x16 bf16): lane L<16 holds N=L, K=k0..k0+15; lanes>=16: K=k0+16..k0+31.
//                  -> one contiguous 32B load (two 16B).
//  C/D (16x16 f32): VGPR r, lane L: row M = r + 8*(L/16), col N = L%16.
__global__ __launch_bounds__(256)
void npair_gemm_exp_kernel(const unsigned short* __restrict__ Ahi,
                           const unsigned short* __restrict__ Alo,
                           const unsigned short* __restrict__ Phi,
                           const unsigned short* __restrict__ Plo,
                           const float* __restrict__ diag,
                           float* __restrict__ rowsum) {
    const int lane    = threadIdx.x & 31;
    const int wave    = threadIdx.x >> 5;        // 8 waves: 2 (M) x 4 (N)
    const int wm      = wave >> 2;               // 0..1
    const int wn      = wave & 3;                // 0..3
    const int i0      = blockIdx.y * BLK_M + wm * (MT * 16);
    const int j0      = blockIdx.x * BLK_N + wn * (NT * 16);
    const int halfSel = lane >> 4;               // 0 | 1
    const int lr      = lane & 15;

    v8f acc[MT][NT];
#pragma unroll
    for (int m = 0; m < MT; ++m)
#pragma unroll
        for (int nt = 0; nt < NT; ++nt) acc[m][nt] = {};

    // Loop-carried pointers; everything else is a compile-time byte offset
    // (m/nt tile stride = 16*DDIM*2 = 32KB < 8MB signed 24-bit ioffset).
    const unsigned short* pAhi = Ahi + (size_t)(i0 + lr) * DDIM + halfSel * 16;
    const unsigned short* pAlo = Alo + (size_t)(i0 + lr) * DDIM + halfSel * 16;
    const unsigned short* pPhi = Phi + (size_t)(j0 + lr) * DDIM + halfSel * 16;
    const unsigned short* pPlo = Plo + (size_t)(j0 + lr) * DDIM + halfSel * 16;

    for (int k0 = 0; k0 < DDIM; k0 += KSTEP) {
        // B fragments for all 4 N-tiles stay register-resident this K-step.
        Frag bh[NT], bl[NT];
#pragma unroll
        for (int nt = 0; nt < NT; ++nt) {
            bh[nt].u[0] = *(const v4u*)(pPhi + nt * 16 * DDIM);
            bh[nt].u[1] = *(const v4u*)(pPhi + nt * 16 * DDIM + 8);
            bl[nt].u[0] = *(const v4u*)(pPlo + nt * 16 * DDIM);
            bl[nt].u[1] = *(const v4u*)(pPlo + nt * 16 * DDIM + 8);
        }
        // Stream A per M-tile; 12 WMMAs per A-fragment pair.
#pragma unroll
        for (int m = 0; m < MT; ++m) {
            Frag ah, al;
            ah.u[0] = *(const v4u*)(pAhi + m * 16 * DDIM);
            ah.u[1] = *(const v4u*)(pAhi + m * 16 * DDIM + 8);
            al.u[0] = *(const v4u*)(pAlo + m * 16 * DDIM);
            al.u[1] = *(const v4u*)(pAlo + m * 16 * DDIM + 8);
#pragma unroll
            for (int nt = 0; nt < NT; ++nt) {
                acc[m][nt] = __builtin_amdgcn_wmma_f32_16x16x32_bf16(
                    false, ah.h, false, bh[nt].h, (short)0, acc[m][nt], false, false);
                acc[m][nt] = __builtin_amdgcn_wmma_f32_16x16x32_bf16(
                    false, ah.h, false, bl[nt].h, (short)0, acc[m][nt], false, false);
                acc[m][nt] = __builtin_amdgcn_wmma_f32_16x16x32_bf16(
                    false, al.h, false, bh[nt].h, (short)0, acc[m][nt], false, false);
            }
        }
        pAhi += KSTEP; pAlo += KSTEP; pPhi += KSTEP; pPlo += KSTEP;
    }

    // Epilogue: e = (j==i) ? 0 : exp(S_ij - diag_i); reduce across the 16
    // lanes holding one row, atomically accumulate into rowsum[i].
#pragma unroll
    for (int m = 0; m < MT; ++m) {
        const int ibase = i0 + 16 * m + halfSel * 8;
#pragma unroll
        for (int nt = 0; nt < NT; ++nt) {
            const int jcol = j0 + 16 * nt + lr;
#pragma unroll
            for (int r = 0; r < 8; ++r) {
                const int i = ibase + r;
                float e = (jcol == i) ? 0.0f : expf(acc[m][nt][r] - diag[i]);
                e += __shfl_xor(e, 1, 32);
                e += __shfl_xor(e, 2, 32);
                e += __shfl_xor(e, 4, 32);
                e += __shfl_xor(e, 8, 32);
                if (lr == 0) atomicAdd(rowsum + i, e);
            }
        }
    }
}

// ---------------- kernel 4: finalize ---------------------------------------
__global__ __launch_bounds__(256)
void npair_finalize_kernel(const float* __restrict__ rowsum,
                           const float* __restrict__ l2acc,
                           float* __restrict__ out) {
    float s = 0.0f;
    for (int i = threadIdx.x; i < NROWS; i += 256) s += log1pf(rowsum[i]);
    __shared__ float sh[256];
    sh[threadIdx.x] = s;
    __syncthreads();
    for (int t = 128; t > 0; t >>= 1) {
        if (threadIdx.x < t) sh[threadIdx.x] += sh[threadIdx.x + t];
        __syncthreads();
    }
    if (threadIdx.x == 0)
        out[0] = sh[0] / (float)NROWS + L2REG * (l2acc[0] / (float)NROWS);
}

// ---------------------------------------------------------------------------
extern "C" void kernel_launch(void* const* d_in, const int* in_sizes, int n_in,
                              void* d_out, int out_size, void* d_ws, size_t ws_size,
                              hipStream_t stream) {
    const float* A = (const float*)d_in[0];   // anchors  (4096 x 1024) f32
    const float* P = (const float*)d_in[1];   // positives(4096 x 1024) f32
    float* out = (float*)d_out;

    const size_t ND = (size_t)NROWS * DDIM;
    // workspace layout (~33.6 MB)
    unsigned short* Ahi = (unsigned short*)d_ws;
    unsigned short* Alo = Ahi + ND;
    unsigned short* Phi = Alo + ND;
    unsigned short* Plo = Phi + ND;
    float* diag   = (float*)(Plo + ND);
    float* rowsum = diag + NROWS;
    float* l2acc  = rowsum + NROWS;

    npair_init_kernel<<<(NROWS + 255) / 256, 256, 0, stream>>>(rowsum, l2acc);
    npair_split_kernel<<<(unsigned)(ND / 256), 256, 0, stream>>>(A, P, Ahi, Alo, Phi, Plo);
    npair_diag_kernel<<<NROWS, 256, 0, stream>>>(A, P, diag, l2acc);
    dim3 grid(NROWS / BLK_N, NROWS / BLK_M);
    npair_gemm_exp_kernel<<<grid, 256, 0, stream>>>(Ahi, Alo, Phi, Plo, diag, rowsum);
    npair_finalize_kernel<<<1, 256, 0, stream>>>(rowsum, l2acc, out);
}
